// MoE_2121713845018
// MI455X (gfx1250) — matrix operand order repaired and checked by reading, not compiled, not run on hardware
//
#include <hip/hip_runtime.h>

// ---------------- problem constants ----------------
#define TT 4096
#define DD 1024
#define HH 2048
#define EE 16
#define ROWCAP 9216             // TT*2 padded per-expert to 64: <= 9200
#define NTILES_MAX (ROWCAP/64)  // 144
#define AP 40                   // padded LDS row stride in halves (64B data + 16B pad)

typedef __attribute__((ext_vector_type(16))) __bf16 v16bf;
typedef __attribute__((ext_vector_type(8)))  float  v8f;
typedef __attribute__((ext_vector_type(4)))  unsigned int u32x4;
typedef __attribute__((ext_vector_type(8)))  unsigned int u32x8;

// ---------------- workspace layout (bytes) ----------------
#define OFF_COUNTS 0u
#define OFF_SEG    64u
#define OFF_META   128u
#define OFF_TILEE  256u
#define OFF_PTOK   1024u
#define OFF_PGATE  (OFF_PTOK + ROWCAP*4u)
#define OFF_TOPI   (OFF_PGATE + ROWCAP*4u)
#define OFF_TOPG   (OFF_TOPI + TT*2u*4u)
#define OFF_XG     (OFF_TOPG + TT*2u*4u)
#define OFF_W1T    (OFF_XG  + (size_t)ROWCAP*DD*2u)
#define OFF_W2T    (OFF_W1T + (size_t)EE*HH*DD*2u)
#define OFF_HWS    (OFF_W2T + (size_t)EE*DD*HH*2u)

#if __has_builtin(__builtin_amdgcn_s_wait_tensorcnt)
#define TDM_WAIT(n) __builtin_amdgcn_s_wait_tensorcnt(n)
#else
#define TDM_WAIT(n) asm volatile("s_wait_tensorcnt %0" :: "i"(n) : "memory")
#endif

__device__ __forceinline__ unsigned short f2bf(float f) {
  unsigned u = __float_as_uint(f);
  unsigned r = u + 0x7FFFu + ((u >> 16) & 1u);   // round to nearest even
  return (unsigned short)(r >> 16);
}

__device__ __forceinline__ v16bf ld_frag(const void* p0, const void* p1) {
  union { uint4 q[2]; v16bf v; } u;
  u.q[0] = *(const uint4*)p0;
  u.q[1] = *(const uint4*)p1;
  return u.v;
}

// ---------------- Tensor Data Mover: 2D tile (rows x 32 halves) -> LDS with 16B row pad ----
// tensor dims == tile dims (always in bounds); stride = matrix row stride (halves).
// data_size=2B (code 1); pad_enable=1, pad_interval code 3 (16 DWORDs=64B), pad_amount code 3 (4 DWORDs=16B).
__device__ __forceinline__ void tdm_load_2d(unsigned lds_addr, const void* gaddr,
                                            unsigned rows, unsigned row_stride_halves) {
  unsigned long long ga = (unsigned long long)gaddr;
  u32x4 g0;
  g0[0] = 0x1u;                                 // count = 1 valid descriptor
  g0[1] = lds_addr;                             // LDS byte address
  g0[2] = (unsigned)ga;                         // global_addr[31:0]
  g0[3] = (unsigned)(ga >> 32) | 0x80000000u;   // global_addr[56:32] | type=2 (bits 127:126)
  u32x8 g1;
  g1[0] = (1u << 16) | (1u << 20) | (3u << 22) | (3u << 25); // data_size|pad_en|interval|amount
  g1[1] = (32u & 0xFFFFu) << 16;                // tensor_dim0 = 32 (lo16 in [31:16])
  g1[2] = 0u | ((rows & 0xFFFFu) << 16);        // tensor_dim0 hi | tensor_dim1 lo16
  g1[3] = (rows >> 16) | (32u << 16);           // tensor_dim1 hi | tile_dim0 = 32
  g1[4] = rows;                                 // tile_dim1; tile_dim2 = 0
  g1[5] = row_stride_halves;                    // tensor_dim0_stride[31:0]
  g1[6] = 0u;
  g1[7] = 0u;
  asm volatile("tensor_load_to_lds %0, %1" :: "s"(g0), "s"(g1) : "memory");
}

// ---------------- init ----------------
__global__ void init_route_kernel(int* counts, int* meta, int* ptok, float* pgate) {
  int i = blockIdx.x * blockDim.x + threadIdx.x;
  if (i < EE) counts[i] = 0;
  if (i < 8)  meta[i] = 0;
  if (i < ROWCAP) { ptok[i] = -1; pgate[i] = 0.0f; }
}

__global__ void zero_out_kernel(float* out, int n) {
  for (int i = blockIdx.x * blockDim.x + threadIdx.x; i < n; i += gridDim.x * blockDim.x)
    out[i] = 0.0f;
}

// ---------------- gating: one wave per token ----------------
__global__ __launch_bounds__(256) void gate_kernel(
    const float* __restrict__ x, const float* __restrict__ eps,
    const float* __restrict__ Wg, const float* __restrict__ Wn,
    int* __restrict__ topi, float* __restrict__ topg, int* __restrict__ counts) {
  const int wid = threadIdx.x >> 5, lane = threadIdx.x & 31;
  const int t = blockIdx.x * 8 + wid;
  const int e = lane & 15;
  const float* w = ((lane < 16) ? Wg : Wn) + (size_t)e * DD;
  const float* xr = x + (size_t)t * DD;
  float acc = 0.0f;
  for (int d = 0; d < DD; d += 4) {
    float4 xv = *(const float4*)(xr + d);
    float4 wv = *(const float4*)(w + d);
    acc += xv.x * wv.x + xv.y * wv.y + xv.z * wv.z + xv.w * wv.w;
  }
  float nz = __shfl(acc, lane + 16, 32);
  float sp = fmaxf(nz, 0.0f) + log1pf(expf(-fabsf(nz)));
  float logit = acc + eps[(size_t)t * EE + e] * sp;    // NOISE_SCALE = 1

  float v = logit; int i = e;
  for (int off = 8; off; off >>= 1) {
    float ov = __shfl_xor(v, off, 16);
    int   oi = __shfl_xor(i, off, 16);
    if (ov > v || (ov == v && oi < i)) { v = ov; i = oi; }
  }
  float m1 = v; int i1 = i;
  float v2 = (e == i1) ? -__builtin_inff() : logit; int j2 = e;
  for (int off = 8; off; off >>= 1) {
    float ov = __shfl_xor(v2, off, 16);
    int   oi = __shfl_xor(j2, off, 16);
    if (ov > v2 || (ov == v2 && oi < j2)) { v2 = ov; j2 = oi; }
  }
  if (lane == 0) {
    float ed = expf(v2 - m1);
    topi[t * 2] = i1; topi[t * 2 + 1] = j2;
    topg[t * 2] = 1.0f / (1.0f + ed); topg[t * 2 + 1] = ed / (1.0f + ed);
    atomicAdd(&counts[i1], 1);
    atomicAdd(&counts[j2], 1);
  }
}

// ---------------- segment scan (64-aligned per expert) ----------------
__global__ void scan_kernel(const int* counts, int* seg, int* meta, int* tile_e) {
  if (threadIdx.x == 0) {
    int acc = 0;
    for (int e = 0; e < EE; e++) {
      seg[e] = acc;
      int c = (counts[e] + 63) & ~63;
      int t0 = acc >> 6, t1 = (acc + c) >> 6;
      for (int t = t0; t < t1; t++) tile_e[t] = e;
      acc += c;
    }
    meta[0] = acc;
    meta[1] = acc >> 6;
  }
}

// ---------------- deterministic compaction: one wave per expert ----------------
__global__ void assign_kernel(const int* __restrict__ topi, const float* __restrict__ topg,
                              const int* __restrict__ seg, int* __restrict__ ptok,
                              float* __restrict__ pgate) {
  const int e = blockIdx.x;
  const int lane = threadIdx.x;
  int base = seg[e], run = 0;
  for (int t0 = 0; t0 < TT; t0 += 32) {
    int t = t0 + lane;
    int a = topi[t * 2], b = topi[t * 2 + 1];
    bool f = (a == e) || (b == e);
    float g = (a == e) ? topg[t * 2] : topg[t * 2 + 1];
    unsigned mask = (unsigned)__ballot(f);
    if (f) {
      int pos = base + run + __popc(mask & ((1u << lane) - 1u));
      ptok[pos] = t; pgate[pos] = g;
    }
    run += __popc(mask);
  }
}

// ---------------- gather routed x rows -> bf16 ----------------
__global__ __launch_bounds__(256) void gather_x_kernel(const float* __restrict__ x,
                                                       const int* __restrict__ ptok,
                                                       const int* __restrict__ meta,
                                                       unsigned short* __restrict__ xg) {
  const int p = blockIdx.x;
  if (p >= meta[0]) return;
  const int tok = ptok[p];
  const int c = threadIdx.x * 4;
  unsigned short o0 = 0, o1 = 0, o2 = 0, o3 = 0;
  if (tok >= 0) {
    float4 v = *(const float4*)(x + (size_t)tok * DD + c);
    o0 = f2bf(v.x); o1 = f2bf(v.y); o2 = f2bf(v.z); o3 = f2bf(v.w);
  }
  ushort4 o; o.x = o0; o.y = o1; o.z = o2; o.w = o3;
  *(ushort4*)(xg + (size_t)p * DD + c) = o;
}

// ---------------- LDS-tiled transpose + f32->bf16: src[e][r][c] -> dst[e][c][r] ----------------
__global__ __launch_bounds__(256) void transpose_bf16(const float* __restrict__ src,
                                                      unsigned short* __restrict__ dst,
                                                      int R, int C) {
  __shared__ float tile[32][33];
  const int e = blockIdx.y;
  const int nx = C >> 5;
  const int r0 = (blockIdx.x / nx) << 5;
  const int c0 = (blockIdx.x % nx) << 5;
  const float* s = src + (size_t)e * R * C;
  unsigned short* d = dst + (size_t)e * R * C;
  const int tx = threadIdx.x & 31, ty = threadIdx.x >> 5;  // 32x8
#pragma unroll
  for (int i = 0; i < 32; i += 8)
    tile[ty + i][tx] = s[(size_t)(r0 + ty + i) * C + (c0 + tx)];
  __syncthreads();
#pragma unroll
  for (int i = 0; i < 32; i += 8)
    d[(size_t)(c0 + ty + i) * R + (r0 + tx)] = f2bf(tile[tx][ty + i]);
}

// ---------------- GEMM1: h = relu(xg @ W1_e + b1_e), TDM double-buffered ----------------
// Block tile 64 rows x 256 cols; 8 waves as 2(wm) x 4(wn); wave tile 32x64 = 8 WMMA/chunk.
__global__ __launch_bounds__(256) void ffn1_kernel(const unsigned short* __restrict__ xg,
                                                   const unsigned short* __restrict__ w1t,
                                                   const float* __restrict__ b1,
                                                   const int* __restrict__ meta,
                                                   const int* __restrict__ tile_e,
                                                   unsigned short* __restrict__ hws) {
  __shared__ unsigned short smA[2][64 * AP];
  __shared__ unsigned short smB[2][256 * AP];
  if ((int)blockIdx.x >= meta[1]) return;
  const int e = tile_e[blockIdx.x];
  const int wid = threadIdx.x >> 5, lane = threadIdx.x & 31;
  const int wm = wid & 1, wn = wid >> 1;
  const int lm = lane & 15, lh = lane >> 4;
  const int rowblk = blockIdx.x * 64;
  const int colblk = blockIdx.y * 256;
  const unsigned short* gA = xg + (size_t)rowblk * DD;
  const unsigned short* gB = w1t + ((size_t)e * HH + colblk) * DD;

  if (wid == 0)      tdm_load_2d((unsigned)(size_t)&smB[0][0], gB, 256, DD);
  else if (wid == 1) tdm_load_2d((unsigned)(size_t)&smA[0][0], gA, 64, DD);

  v8f acc[2][4] = {};
  const int NK = DD / 32;
  for (int kc = 0; kc < NK; ++kc) {
    const int cur = kc & 1;
    if (kc + 1 < NK) {
      if (wid == 0)      tdm_load_2d((unsigned)(size_t)&smB[cur ^ 1][0], gB + (kc + 1) * 32, 256, DD);
      else if (wid == 1) tdm_load_2d((unsigned)(size_t)&smA[cur ^ 1][0], gA + (kc + 1) * 32, 64, DD);
      if (wid < 2) TDM_WAIT(1);
    } else {
      if (wid < 2) TDM_WAIT(0);
    }
    __syncthreads();
    v16bf a[2], b[4];
#pragma unroll
    for (int mi = 0; mi < 2; mi++) {
      const unsigned short* aL = &smA[cur][(wm * 32 + mi * 16 + lm) * AP + lh * 8];
      a[mi] = ld_frag(aL, aL + 16);
    }
#pragma unroll
    for (int j = 0; j < 4; j++) {
      const unsigned short* bL = &smB[cur][(wn * 64 + j * 16 + lm) * AP + lh * 8];
      b[j] = ld_frag(bL, bL + 16);
    }
#pragma unroll
    for (int mi = 0; mi < 2; mi++)
#pragma unroll
      for (int j = 0; j < 4; j++)
        acc[mi][j] = __builtin_amdgcn_wmma_f32_16x16x32_bf16(false, a[mi], false, b[j],
                                                             (short)0, acc[mi][j], false, false);
    __syncthreads();
  }
#pragma unroll
  for (int mi = 0; mi < 2; mi++)
#pragma unroll
    for (int j = 0; j < 4; j++) {
      const int col = colblk + wn * 64 + j * 16 + lm;
      const float bias = b1[e * HH + col];
#pragma unroll
      for (int r = 0; r < 8; r++) {
        const int row = rowblk + wm * 32 + mi * 16 + r + lh * 8;
        hws[(size_t)row * HH + col] = f2bf(fmaxf(acc[mi][j][r] + bias, 0.0f));
      }
    }
}

// ---------------- GEMM2: y = h @ W2_e + b2_e, scaled scatter-add, TDM double-buffered ----------
__global__ __launch_bounds__(256) void ffn2_kernel(const unsigned short* __restrict__ hws,
                                                   const unsigned short* __restrict__ w2t,
                                                   const float* __restrict__ b2,
                                                   const int* __restrict__ meta,
                                                   const int* __restrict__ tile_e,
                                                   const int* __restrict__ ptok,
                                                   const float* __restrict__ pgate,
                                                   float* __restrict__ out) {
  __shared__ unsigned short smA[2][64 * AP];
  __shared__ unsigned short smB[2][256 * AP];
  if ((int)blockIdx.x >= meta[1]) return;
  const int e = tile_e[blockIdx.x];
  const int wid = threadIdx.x >> 5, lane = threadIdx.x & 31;
  const int wm = wid & 1, wn = wid >> 1;
  const int lm = lane & 15, lh = lane >> 4;
  const int rowblk = blockIdx.x * 64;
  const int colblk = blockIdx.y * 256;
  const unsigned short* gA = hws + (size_t)rowblk * HH;
  const unsigned short* gB = w2t + ((size_t)e * DD + colblk) * HH;

  if (wid == 0)      tdm_load_2d((unsigned)(size_t)&smB[0][0], gB, 256, HH);
  else if (wid == 1) tdm_load_2d((unsigned)(size_t)&smA[0][0], gA, 64, HH);

  v8f acc[2][4] = {};
  const int NK = HH / 32;
  for (int kc = 0; kc < NK; ++kc) {
    const int cur = kc & 1;
    if (kc + 1 < NK) {
      if (wid == 0)      tdm_load_2d((unsigned)(size_t)&smB[cur ^ 1][0], gB + (kc + 1) * 32, 256, HH);
      else if (wid == 1) tdm_load_2d((unsigned)(size_t)&smA[cur ^ 1][0], gA + (kc + 1) * 32, 64, HH);
      if (wid < 2) TDM_WAIT(1);
    } else {
      if (wid < 2) TDM_WAIT(0);
    }
    __syncthreads();
    v16bf a[2], b[4];
#pragma unroll
    for (int mi = 0; mi < 2; mi++) {
      const unsigned short* aL = &smA[cur][(wm * 32 + mi * 16 + lm) * AP + lh * 8];
      a[mi] = ld_frag(aL, aL + 16);
    }
#pragma unroll
    for (int j = 0; j < 4; j++) {
      const unsigned short* bL = &smB[cur][(wn * 64 + j * 16 + lm) * AP + lh * 8];
      b[j] = ld_frag(bL, bL + 16);
    }
#pragma unroll
    for (int mi = 0; mi < 2; mi++)
#pragma unroll
      for (int j = 0; j < 4; j++)
        acc[mi][j] = __builtin_amdgcn_wmma_f32_16x16x32_bf16(false, a[mi], false, b[j],
                                                             (short)0, acc[mi][j], false, false);
    __syncthreads();
  }
  int   toks[2][8];
  float gts[2][8];
#pragma unroll
  for (int mi = 0; mi < 2; mi++)
#pragma unroll
    for (int r = 0; r < 8; r++) {
      const int row = rowblk + wm * 32 + mi * 16 + r + lh * 8;
      toks[mi][r] = ptok[row];
      gts[mi][r] = pgate[row];
    }
#pragma unroll
  for (int mi = 0; mi < 2; mi++)
#pragma unroll
    for (int j = 0; j < 4; j++) {
      const int col = colblk + wn * 64 + j * 16 + lm;
      const float bias = b2[e * DD + col];
#pragma unroll
      for (int r = 0; r < 8; r++) {
        if (toks[mi][r] >= 0) {
          float val = (acc[mi][j][r] + bias) * gts[mi][r];
          unsafeAtomicAdd(out + (size_t)toks[mi][r] * DD + col, val);
        }
      }
    }
}

// ---------------- launch ----------------
extern "C" void kernel_launch(void* const* d_in, const int* in_sizes, int n_in,
                              void* d_out, int out_size, void* d_ws, size_t ws_size,
                              hipStream_t stream) {
  (void)in_sizes; (void)n_in; (void)out_size; (void)ws_size;
  const float* x   = (const float*)d_in[0];
  const float* eps = (const float*)d_in[1];
  const float* Wg  = (const float*)d_in[2];
  const float* Wn  = (const float*)d_in[3];
  const float* W1  = (const float*)d_in[4];
  const float* b1  = (const float*)d_in[5];
  const float* W2  = (const float*)d_in[6];
  const float* b2  = (const float*)d_in[7];
  float* out = (float*)d_out;

  char* w = (char*)d_ws;
  int*   counts = (int*)(w + OFF_COUNTS);
  int*   seg    = (int*)(w + OFF_SEG);
  int*   meta   = (int*)(w + OFF_META);
  int*   tile_e = (int*)(w + OFF_TILEE);
  int*   ptok   = (int*)(w + OFF_PTOK);
  float* pgate  = (float*)(w + OFF_PGATE);
  int*   topi   = (int*)(w + OFF_TOPI);
  float* topg   = (float*)(w + OFF_TOPG);
  unsigned short* xg  = (unsigned short*)(w + OFF_XG);
  unsigned short* w1t = (unsigned short*)(w + OFF_W1T);
  unsigned short* w2t = (unsigned short*)(w + OFF_W2T);
  unsigned short* hws = (unsigned short*)(w + OFF_HWS);

  init_route_kernel<<<(ROWCAP + 255) / 256, 256, 0, stream>>>(counts, meta, ptok, pgate);
  zero_out_kernel<<<1024, 256, 0, stream>>>(out, TT * DD);
  gate_kernel<<<TT / 8, 256, 0, stream>>>(x, eps, Wg, Wn, topi, topg, counts);
  scan_kernel<<<1, 32, 0, stream>>>(counts, seg, meta, tile_e);
  assign_kernel<<<EE, 32, 0, stream>>>(topi, topg, seg, ptok, pgate);
  gather_x_kernel<<<ROWCAP, 256, 0, stream>>>(x, ptok, meta, xg);
  transpose_bf16<<<dim3((DD / 32) * (HH / 32), EE), 256, 0, stream>>>(W1, w1t, DD, HH);
  transpose_bf16<<<dim3((HH / 32) * (DD / 32), EE), 256, 0, stream>>>(W2, w2t, HH, DD);
  ffn1_kernel<<<dim3(NTILES_MAX, HH / 256), 256, 0, stream>>>(xg, w1t, b1, meta, tile_e, hws);
  ffn2_kernel<<<dim3(NTILES_MAX, DD / 256), 256, 0, stream>>>(hws, w2t, b2, meta, tile_e, ptok, pgate, out);
}
